// GRNN_Cell_41515153883224
// MI455X (gfx1250) — matrix operand order repaired and checked by reading, not compiled, not run on hardware
//
#include <hip/hip_runtime.h>
#include <hip/hip_bf16.h>

typedef __attribute__((ext_vector_type(16))) _Float16 v16h;
typedef __attribute__((ext_vector_type(8)))  float    v8f;

#define F_IN 128
#define C_OUT 64
#define D_IN 192
#define NEG_SLOPE 0.2f

// ---------------------------------------------------------------------------
// WMMA GEMM:  h = [X | H (optionally * R)] @ W      ([N,192] x [192,64])
// One wave computes a 16-row x 64-col stripe (4 16x16 f32 accum tiles).
// A/B fragments converted to f16 on the fly per CDNA5 ISA 7.12.2 layouts.
// ---------------------------------------------------------------------------
__global__ void gat_gemm_wmma(const float* __restrict__ X,
                              const float* __restrict__ H,
                              const float* __restrict__ R,   // used iff useR
                              const float* __restrict__ W,   // [192,64] row major
                              float* __restrict__ out,       // [N,64]
                              int N, int useR)
{
    const int wave = threadIdx.x >> 5;
    const int lane = threadIdx.x & 31;
    const int row0 = (blockIdx.x * (blockDim.x >> 5) + wave) * 16;
    if (row0 >= N) return;                    // uniform per wave

    const int m    = lane & 15;               // row (A) / col (B,D) within tile
    const int half = lane >> 4;               // lane group 0/1
    const int row  = row0 + m;
    const bool rowOK = (row < N);
    const int kbaseA = half * 8;              // A: lanes 0-15 -> K 0..7/16..23 ; 16-31 -> K 8..15/24..31
    const int kbaseB = half * 16;             // unused; B fragment takes the same k order as A (kbaseA)

    v8f acc0 = {}, acc1 = {}, acc2 = {}, acc3 = {};

    for (int kk = 0; kk < D_IN; kk += 32) {
        // ---- A fragment (16x32 f16, one row per lane-position) ----
        v16h afrag;
#pragma unroll
        for (int i = 0; i < 16; ++i) {
            const int f = kk + ((i < 8) ? (kbaseA + i) : (16 + kbaseA + (i - 8)));
            float v = 0.0f;
            if (rowOK) {
                if (f < F_IN) {
                    v = X[row * F_IN + f];
                } else {
                    v = H[row * C_OUT + (f - F_IN)];
                    if (useR) v *= R[row * C_OUT + (f - F_IN)];
                }
            }
            afrag[i] = (_Float16)v;
        }
        // ---- 4 column tiles ----
#pragma unroll
        for (int t = 0; t < 4; ++t) {
            const int col = t * 16 + m;
            v16h bfrag;
#pragma unroll
            for (int i = 0; i < 16; ++i) {
                const int k = kk + ((i < 8) ? (kbaseA + i) : (16 + kbaseA + (i - 8)));
                bfrag[i] = (_Float16)W[k * C_OUT + col];
            }
            v8f c = (t == 0) ? acc0 : (t == 1) ? acc1 : (t == 2) ? acc2 : acc3;
            c = __builtin_amdgcn_wmma_f32_16x16x32_f16(
                    false, afrag, false, bfrag, (short)0, c, false, false);
            if      (t == 0) acc0 = c;
            else if (t == 1) acc1 = c;
            else if (t == 2) acc2 = c;
            else             acc3 = c;
        }
    }

    // D layout: lane holds column m, VGPR v -> row v + half*8
#pragma unroll
    for (int t = 0; t < 4; ++t) {
        const int col = t * 16 + m;
        v8f c = (t == 0) ? acc0 : (t == 1) ? acc1 : (t == 2) ? acc2 : acc3;
#pragma unroll
        for (int v = 0; v < 8; ++v) {
            const int r = row0 + v + half * 8;
            if (r < N) out[r * C_OUT + col] = c[v];
        }
    }
}

// ---------------------------------------------------------------------------
// Per-node attention logits: al_s[n] = <h[n,:], a_src>, al_d[n] = <h[n,:], a_dst>
// Wave per node, shuffle reduction (wave32).
// ---------------------------------------------------------------------------
__global__ void gat_logits(const float* __restrict__ h,
                           const float* __restrict__ asrc,
                           const float* __restrict__ adst,
                           float* __restrict__ als, float* __restrict__ ald, int N)
{
    const int wave = threadIdx.x >> 5;
    const int lane = threadIdx.x & 31;
    const int node = blockIdx.x * (blockDim.x >> 5) + wave;
    if (node >= N) return;                    // uniform per wave
    float s = 0.f, d = 0.f;
#pragma unroll
    for (int j = 0; j < 2; ++j) {
        const int c = lane + 32 * j;
        const float hv = h[node * C_OUT + c];
        s += hv * asrc[c];
        d += hv * adst[c];
    }
#pragma unroll
    for (int off = 16; off > 0; off >>= 1) {
        s += __shfl_down(s, off, 32);
        d += __shfl_down(d, off, 32);
    }
    if (lane == 0) { als[node] = s; ald[node] = d; }
}

// order-preserving float <-> uint key for atomicMax
__device__ __forceinline__ unsigned f2key(float f) {
    unsigned b = __float_as_uint(f);
    return (b & 0x80000000u) ? ~b : (b | 0x80000000u);
}
__device__ __forceinline__ float key2f(unsigned k) {
    return __uint_as_float((k & 0x80000000u) ? (k & 0x7FFFFFFFu) : ~k);
}

__device__ __forceinline__ float lrelu(float a) {
    return (a > 0.f) ? a : a * NEG_SLOPE;
}

// ---------------------------------------------------------------------------
// Edge pass 1: per-dst max of leaky_relu(al_s[src] + al_d[dst])
// ---------------------------------------------------------------------------
__global__ void gat_edge_max(const int* __restrict__ srcs, const int* __restrict__ dsts,
                             int E, int N,
                             const float* __restrict__ als, const float* __restrict__ ald,
                             unsigned* __restrict__ maxkey)
{
    const int e = blockIdx.x * blockDim.x + threadIdx.x;
    const int total = E + N;
    if (e >= total) return;
    const int s = (e < E) ? srcs[e] : (e - E);   // trailing self loops
    const int d = (e < E) ? dsts[e] : (e - E);
    const float a = lrelu(als[s] + ald[d]);
    atomicMax(&maxkey[d], f2key(a));
}

// ---------------------------------------------------------------------------
// Edge pass 2: w = exp(a - max[dst]); denom[dst] += w; agg[dst,:] += h[src,:]*w
// Wave per edge, each lane owns 2 channels -> coalesced 256B dst lines.
// ---------------------------------------------------------------------------
__global__ void gat_edge_accum(const int* __restrict__ srcs, const int* __restrict__ dsts,
                               int E, int N,
                               const float* __restrict__ als, const float* __restrict__ ald,
                               const unsigned* __restrict__ maxkey,
                               const float* __restrict__ h,
                               float* __restrict__ den, float* __restrict__ agg)
{
    const int wave = threadIdx.x >> 5;
    const int lane = threadIdx.x & 31;
    const int e = blockIdx.x * (blockDim.x >> 5) + wave;
    const int total = E + N;
    if (e >= total) return;                   // uniform per wave
    const int s = (e < E) ? srcs[e] : (e - E);
    const int d = (e < E) ? dsts[e] : (e - E);
    const float a = lrelu(als[s] + ald[d]);
    const float w = expf(a - key2f(maxkey[d]));
    if (lane == 0) atomicAdd(&den[d], w);
#pragma unroll
    for (int j = 0; j < 2; ++j) {
        const int c = lane + 32 * j;
        atomicAdd(&agg[d * C_OUT + c], h[s * C_OUT + c] * w);
    }
}

// ---------------------------------------------------------------------------
// In-place: gate = sigmoid(agg / (denom + eps) + bias)
// ---------------------------------------------------------------------------
__global__ void gat_norm_sigmoid(float* __restrict__ agg, const float* __restrict__ den,
                                 const float* __restrict__ bias, int N)
{
    const int i = blockIdx.x * blockDim.x + threadIdx.x;
    if (i >= N * C_OUT) return;
    const int n = i >> 6, c = i & 63;
    const float v = agg[i] / (den[n] + 1e-16f) + bias[c];
    agg[i] = 1.f / (1.f + expf(-v));
}

// ---------------------------------------------------------------------------
// out = Z*H + (1-Z)*tanh(agg_h/(den_h+eps) + b_h)
// ---------------------------------------------------------------------------
__global__ void gat_final(const float* __restrict__ aggh, const float* __restrict__ denh,
                          const float* __restrict__ bh, const float* __restrict__ Z,
                          const float* __restrict__ H, float* __restrict__ out, int N)
{
    const int i = blockIdx.x * blockDim.x + threadIdx.x;
    if (i >= N * C_OUT) return;
    const int n = i >> 6, c = i & 63;
    const float ht = tanhf(aggh[i] / (denh[n] + 1e-16f) + bh[c]);
    const float z = Z[i];
    out[i] = z * H[i] + (1.f - z) * ht;
}

// ---------------------------------------------------------------------------
extern "C" void kernel_launch(void* const* d_in, const int* in_sizes, int n_in,
                              void* d_out, int out_size, void* d_ws, size_t ws_size,
                              hipStream_t stream)
{
    const float* X   = (const float*)d_in[0];
    const int*   ei  = (const int*)  d_in[1];
    const float* H   = (const float*)d_in[2];
    const float* Wz  = (const float*)d_in[3];
    const float* azs = (const float*)d_in[4];
    const float* azd = (const float*)d_in[5];
    const float* bz  = (const float*)d_in[6];
    const float* Wr  = (const float*)d_in[7];
    const float* ars = (const float*)d_in[8];
    const float* ard = (const float*)d_in[9];
    const float* br  = (const float*)d_in[10];
    const float* Wh  = (const float*)d_in[11];
    const float* ahs = (const float*)d_in[12];
    const float* ahd = (const float*)d_in[13];
    const float* bh  = (const float*)d_in[14];

    const int N = in_sizes[0] / F_IN;
    const int E = in_sizes[1] / 2;
    const int* src = ei;
    const int* dst = ei + E;

    const size_t N64 = (size_t)N * C_OUT;

    // workspace partition
    float* hz   = (float*)d_ws;
    float* hr   = hz + N64;
    float* hh   = hr + N64;
    float* alsz = hh + N64;
    float* aldz = alsz + N;
    float* alsr = aldz + N;
    float* aldr = alsr + N;
    float* alsh = aldr + N;
    float* aldh = alsh + N;
    // zero-initialized region starts here
    float*    zero0 = aldh + N;
    float*    aggz  = zero0;            // becomes Z in place
    float*    aggr  = aggz + N64;       // becomes R in place
    float*    aggh  = aggr + N64;
    float*    denz  = aggh + N64;
    float*    denr  = denz + N;
    float*    denh  = denr + N;
    unsigned* maxz  = (unsigned*)(denh + N);
    unsigned* maxr  = maxz + N;
    unsigned* maxh  = maxr + N;
    const size_t zero_bytes = (3 * N64 + 6 * (size_t)N) * sizeof(float);
    const size_t need_bytes = (6 * N64 + 12 * (size_t)N) * sizeof(float);
    if (need_bytes > ws_size) return;   // insufficient scratch: do nothing (deterministic)

    (void)hipMemsetAsync(zero0, 0, zero_bytes, stream);

    const int total   = E + N;
    const int gemmBlk = 128;                                 // 4 waves -> 64 rows / block
    const int gemmGrd = (N + 63) / 64;
    const int nodeGrd = (N + 7) / 8;                         // 8 waves / block, wave per node
    const int edgeGrd = (total + 255) / 256;
    const int waveGrd = (total + 7) / 8;                     // wave per edge
    const int elemGrd = ((int)N64 + 255) / 256;

    // ---- gates z, r : h = [X|H] @ W ----
    gat_gemm_wmma<<<gemmGrd, gemmBlk, 0, stream>>>(X, H, nullptr, Wz, hz, N, 0);
    gat_gemm_wmma<<<gemmGrd, gemmBlk, 0, stream>>>(X, H, nullptr, Wr, hr, N, 0);
    gat_logits<<<nodeGrd, 256, 0, stream>>>(hz, azs, azd, alsz, aldz, N);
    gat_logits<<<nodeGrd, 256, 0, stream>>>(hr, ars, ard, alsr, aldr, N);
    gat_edge_max<<<edgeGrd, 256, 0, stream>>>(src, dst, E, N, alsz, aldz, maxz);
    gat_edge_max<<<edgeGrd, 256, 0, stream>>>(src, dst, E, N, alsr, aldr, maxr);
    gat_edge_accum<<<waveGrd, 256, 0, stream>>>(src, dst, E, N, alsz, aldz, maxz, hz, denz, aggz);
    gat_edge_accum<<<waveGrd, 256, 0, stream>>>(src, dst, E, N, alsr, aldr, maxr, hr, denr, aggr);
    gat_norm_sigmoid<<<elemGrd, 256, 0, stream>>>(aggz, denz, bz, N);   // aggz -> Z
    gat_norm_sigmoid<<<elemGrd, 256, 0, stream>>>(aggr, denr, br, N);   // aggr -> R

    // ---- gate h : h = [X | H*R] @ Wh ----
    gat_gemm_wmma<<<gemmGrd, gemmBlk, 0, stream>>>(X, H, aggr, Wh, hh, N, 1);
    gat_logits<<<nodeGrd, 256, 0, stream>>>(hh, ahs, ahd, alsh, aldh, N);
    gat_edge_max<<<edgeGrd, 256, 0, stream>>>(src, dst, E, N, alsh, aldh, maxh);
    gat_edge_accum<<<waveGrd, 256, 0, stream>>>(src, dst, E, N, alsh, aldh, maxh, hh, denh, aggh);

    // ---- combine ----
    gat_final<<<elemGrd, 256, 0, stream>>>(aggh, denh, bh, aggz, H, (float*)d_out, N);
}
